// CrossUserAttention_72713796321807
// MI455X (gfx1250) — matrix operand order repaired and checked by reading, not compiled, not run on hardware
//
#include <hip/hip_runtime.h>
#include <math.h>
#include <stdint.h>

// ---------------- types / WMMA helpers ----------------
typedef _Float16 f16;
typedef __attribute__((ext_vector_type(16))) _Float16 v16h;
typedef __attribute__((ext_vector_type(8)))  float    v8f;
typedef __attribute__((ext_vector_type(4)))  float    vf4;

union Frag {
  v16h v;
  vf4  q[2];
};

__device__ __forceinline__ v8f wmma16(const Frag& a, const Frag& b, v8f c) {
  // D = A(16x32 f16) * B(32x16 f16) + C(16x16 f32)
  return __builtin_amdgcn_wmma_f32_16x16x32_f16(false, a.v, false, b.v,
                                                (short)0, c, false, false);
}

__device__ __forceinline__ float wave_sum32(float x) {
#pragma unroll
  for (int m = 16; m >= 1; m >>= 1) x += __shfl_xor(x, m, 32);
  return x;
}

// ---- CDNA5 async Global->LDS copy (ASYNCcnt-tracked, ISA §10 / §15.18.3) ----
// One instruction per wave: every lane copies 16B from its global address to
// its LDS byte-offset. vdst = LDS offset VGPR, vaddr = 64-bit address pair.
__device__ __forceinline__ void async_copy_b128(uint32_t lds_off, const void* gsrc) {
  asm volatile("global_load_async_to_lds_b128 %0, %1, off"
               :: "v"(lds_off), "v"(gsrc) : "memory");
}
template <int N>
__device__ __forceinline__ void wait_asynccnt() {
  asm volatile("s_wait_asynccnt %0" :: "i"(N) : "memory");
}

// ---------------- problem constants ----------------
constexpr int CD = 256;     // channels
constexpr int SD = 1024;    // tokens (32*32)
constexpr size_t OUT_SHARED = (size_t)16 * CD * SD; // 4194304
constexpr size_t OUT_GATE   = OUT_SHARED;
constexpr size_t OUT_SIM    = OUT_SHARED + 8;

// ---------------- workspace layout (bytes) ----------------
constexpr size_t OFF_WIN  = 0;                               // in_proj^T  f16 [256][768]
constexpr size_t OFF_WOUT = OFF_WIN  + (size_t)196608 * 2;   // out_proj^T f16 [256][256]
constexpr size_t OFF_W1   = OFF_WOUT + (size_t)65536  * 2;   // w1^T       f16 [256][1024]
constexpr size_t OFF_W2   = OFF_W1   + (size_t)262144 * 2;   // w2^T       f16 [1024][256]
constexpr size_t OFF_MSUM = OFF_W2   + (size_t)262144 * 2;   // f32 [16][256]
constexpr size_t OFF_GVAL = OFF_MSUM + (size_t)16*256*4;     // f32 [8] (res_scale*gate)
constexpr size_t OFF_QLN  = OFF_GVAL + 256;                  // f16 [16][1024][256]
constexpr size_t OFF_QT   = OFF_QLN  + (size_t)8388608;      // f16 [16][8][32][1024] (Q^T)
constexpr size_t OFF_KB   = OFF_QT   + (size_t)8388608;      // f16 [16][1024][256]   (K)
constexpr size_t OFF_VB   = OFF_KB   + (size_t)8388608;      // f16 [16][1024][256]   (V)
constexpr size_t OFF_CTX  = OFF_VB   + (size_t)8388608;      // f16 [16][1024][256]
constexpr size_t OFF_NEWX = OFF_CTX  + (size_t)8388608;      // f32 [16][1024][256]
constexpr size_t OFF_HLN  = OFF_NEWX + (size_t)16777216;     // f16 [16][1024][256]
constexpr size_t OFF_T1   = OFF_QLN;                         // overlay: f16 [16][1024][1024]
// total ~68.7 MB

// ---------------- prep: transpose weights -> K-major f16 ----------------
__global__ __launch_bounds__(256) void k_prep(const float* __restrict__ inw,
                                              const float* __restrict__ outw,
                                              const float* __restrict__ w1,
                                              const float* __restrict__ w2,
                                              char* __restrict__ ws) {
  f16* winT  = (f16*)(ws + OFF_WIN);
  f16* woutT = (f16*)(ws + OFF_WOUT);
  f16* w1T   = (f16*)(ws + OFF_W1);
  f16* w2T   = (f16*)(ws + OFF_W2);
  for (unsigned id = blockIdx.x * 256u + threadIdx.x; id < 786432u;
       id += gridDim.x * 256u) {
    if (id < 196608u) {            // in_proj (768,256) -> (256,768)
      int n = id % 768, k = id / 768;
      winT[k * 768 + n] = (f16)inw[n * 256 + k];
    } else if (id < 262144u) {     // out_proj (256,256) -> (256,256)
      unsigned i = id - 196608u; int n = i % 256, k = i / 256;
      woutT[k * 256 + n] = (f16)outw[n * 256 + k];
    } else if (id < 524288u) {     // w1 (1024,256) -> (256,1024)
      unsigned i = id - 262144u; int n = i % 1024, k = i / 1024;
      w1T[k * 1024 + n] = (f16)w1[n * 256 + k];
    } else {                       // w2 (256,1024) -> (1024,256)
      unsigned i = id - 524288u; int n = i % 256, k = i / 256;
      w2T[k * 256 + n] = (f16)w2[n * 1024 + k];
    }
  }
}

// ---------------- copy z_sem -> d_out (baseline for non-updated imgs) -------
__global__ __launch_bounds__(256) void k_copy(const float* __restrict__ src,
                                              float* __restrict__ dst) {
  size_t i = (size_t)blockIdx.x * 256 + threadIdx.x;
  ((vf4*)dst)[i] = ((const vf4*)src)[i];
}

// ---------------- LN of base tokens (channel-major source) ----------------
__global__ __launch_bounds__(256) void k_ln(const float* __restrict__ zsem,
                                            const int* __restrict__ pidx,
                                            const float* __restrict__ lnw,
                                            const float* __restrict__ lnb,
                                            f16* __restrict__ qln) {
  __shared__ float tile[256 * 33]; // [c][s_local], padded
  const int t  = blockIdx.y;
  const int s0 = blockIdx.x * 32;
  const int img = pidx[t];
  const float* zp = zsem + (size_t)img * CD * SD;
#pragma unroll 4
  for (int r = 0; r < 32; ++r) {
    int idx = threadIdx.x + 256 * r;
    int c = idx >> 5, sl = idx & 31;
    tile[c * 33 + sl] = zp[(size_t)c * SD + s0 + sl];
  }
  __syncthreads();
  const int w = threadIdx.x >> 5, lane = threadIdx.x & 31;
#pragma unroll
  for (int u = 0; u < 4; ++u) {
    int sl = w * 4 + u;
    float v[8]; float sum = 0.f, sq = 0.f;
#pragma unroll
    for (int k = 0; k < 8; ++k) {
      v[k] = tile[(lane + 32 * k) * 33 + sl];
      sum += v[k]; sq += v[k] * v[k];
    }
    sum = wave_sum32(sum); sq = wave_sum32(sq);
    float mean = sum * (1.f / 256.f);
    float var  = sq * (1.f / 256.f) - mean * mean;
    float rstd = rsqrtf(var + 1e-5f);
    f16* op = qln + ((size_t)t * SD + s0 + sl) * CD;
#pragma unroll
    for (int k = 0; k < 8; ++k) {
      int c = lane + 32 * k;
      op[c] = (f16)(((v[k] - mean) * rstd) * lnw[c] + lnb[c]);
    }
  }
}

// ---------------- token-mean of LN output ----------------
__global__ __launch_bounds__(256) void k_mean(const f16* __restrict__ qln,
                                              float* __restrict__ msum) {
  int t = blockIdx.x, c = threadIdx.x;
  const f16* p = qln + (size_t)t * SD * CD + c;
  float s = 0.f;
  for (int i = 0; i < SD; ++i) s += (float)p[(size_t)i * CD];
  msum[t * 256 + c] = s * (1.f / 1024.f);
}

// ---------------- cosine gate ----------------
__global__ __launch_bounds__(256) void k_gate(char* __restrict__ ws,
                                              const float* __restrict__ gate_scale,
                                              const float* __restrict__ res_scale,
                                              float* __restrict__ dout) {
  __shared__ float r0[256], r1[256], r2[256];
  int p = blockIdx.x, c = threadIdx.x;
  const float* msum = (const float*)(ws + OFF_MSUM);
  float mi = msum[(2 * p) * 256 + c];
  float mj = msum[(2 * p + 1) * 256 + c];
  r0[c] = mi * mj; r1[c] = mi * mi; r2[c] = mj * mj;
  __syncthreads();
  for (int st = 128; st > 0; st >>= 1) {
    if (c < st) { r0[c] += r0[c + st]; r1[c] += r1[c + st]; r2[c] += r2[c + st]; }
    __syncthreads();
  }
  if (c == 0) {
    float nmi = sqrtf(r1[0]), nmj = sqrtf(r2[0]);
    float sim = r0[0] / (fmaxf(nmi, 1e-6f) * fmaxf(nmj, 1e-6f));
    float gate = 1.f / (1.f + __expf(-gate_scale[0] * sim));
    dout[OUT_GATE + p] = gate;
    dout[OUT_SIM + p]  = sim;
    ((float*)(ws + OFF_GVAL))[p] = res_scale[0] * gate;
  }
}

// ---------------- generic WMMA GEMM: C(M,N) = A(M,K) @ Wt(K,N) + bias --------
// B k-tiles (32x64, shared by all 8 waves) flow through a 4-deep LDS pipeline
// fed by the CDNA5 async Global->LDS engine, stage-ahead = 2, ONE barrier per
// k-step: s_wait_asynccnt<=2 before the barrier proves the current buffer is
// complete in every wave, and stage-ahead 2 means the buffer written at step k
// was last read at step k-2 (two barriers ago) -> no WAR hazard.
constexpr int EP_QKV = 0, EP_OUT = 1, EP_MLP1 = 2, EP_MLP2 = 3;

template <int EP, int M, int N, int K>
__global__ __launch_bounds__(256) void k_gemm(const f16* __restrict__ Abase,
                                              const f16* __restrict__ Bt,
                                              const float* __restrict__ bias,
                                              char* __restrict__ ws,
                                              const float* __restrict__ zsem,
                                              const int* __restrict__ pidx,
                                              float* __restrict__ dout) {
  constexpr int LDB = 72;                       // padded row stride (halves, 144B)
  constexpr int NK  = K / 32;                   // 8 or 32 (multiple of 4)
  __shared__ __align__(16) f16 Bs[4][32 * LDB];
  const int t    = blockIdx.z;
  const int m0   = blockIdx.y * 128 + (threadIdx.x >> 5) * 16;
  const int n0   = blockIdx.x * 64;
  const int lane = threadIdx.x & 31;
  const int half = lane >> 4, lm = lane & 15;
  const f16* A  = Abase + (size_t)t * M * K;
  const f16* Ar = A + (size_t)(m0 + lm) * K + 8 * half;

  // each thread owns one 16B chunk of the 32x64 B k-tile
  const int br = threadIdx.x >> 3;              // 0..31 (k row within tile)
  const int bc = (threadIdx.x & 7) * 8;         // 0,8,...,56 (halves)
  const f16* bsrc = Bt + (size_t)br * N + n0 + bc;
  uint32_t bd[4];
#pragma unroll
  for (int b = 0; b < 4; ++b)
    bd[b] = (uint32_t)(uintptr_t)&Bs[b][br * LDB + bc];

  async_copy_b128(bd[0], bsrc);                         // stage ks=0
  async_copy_b128(bd[1], bsrc + (size_t)32 * N);        // stage ks=1

  v8f acc[4] = {};
  auto compute = [&](int ks, int buf) {
    Frag a;
    const f16* apg = Ar + ks * 32;
    a.q[0] = *(const vf4*)apg;
    a.q[1] = *(const vf4*)(apg + 16);
    if (ks + 1 < NK) __builtin_prefetch(apg + 32, 0, 3);
    const f16* bl = &Bs[buf][lm];               // B-frag col n = lane%16, k over elements
#pragma unroll
    for (int j = 0; j < 4; ++j) {
      Frag b;
#pragma unroll
      for (int s = 0; s < 16; ++s)
        b.v[s] = bl[(s < 8 ? 8 * half + s : 8 + 8 * half + s) * LDB + 16 * j];
      acc[j] = wmma16(a, b, acc[j]);
    }
  };

  for (int ko = 0; ko < NK - 4; ko += 4) {      // steady state
#pragma unroll
    for (int ki = 0; ki < 4; ++ki) {
      const int ks = ko + ki;
      async_copy_b128(bd[(ki + 2) & 3], bsrc + (size_t)(ks + 2) * 32 * N);
      wait_asynccnt<2>();                       // stage ks complete (in order)
      __syncthreads();                          // ... in every wave
      compute(ks, ki);
    }
  }
#pragma unroll
  for (int ki = 0; ki < 4; ++ki) {              // last 4 k-steps
    const int ks = NK - 4 + ki;
    if (ki < 2) {
      async_copy_b128(bd[(ki + 2) & 3], bsrc + (size_t)(ks + 2) * 32 * N);
      wait_asynccnt<2>();
    } else if (ki == 2) {
      wait_asynccnt<1>();
    } else {
      wait_asynccnt<0>();
    }
    __syncthreads();
    compute(ks, ki);
  }

  f16* qt  = (f16*)(ws + OFF_QT);
  f16* kb  = (f16*)(ws + OFF_KB);
  f16* vb  = (f16*)(ws + OFF_VB);
  float* newx = (float*)(ws + OFF_NEWX);
  f16* t1 = (f16*)(ws + OFF_T1);

#pragma unroll
  for (int j = 0; j < 4; ++j) {
#pragma unroll
    for (int i = 0; i < 8; ++i) {
      int m = m0 + i + 8 * half;          // C-frag: M = vgpr + 8*(lane/16)
      int n = n0 + 16 * j + lm;           //         N = lane%16
      float val = acc[j][i] + bias[n];
      if constexpr (EP == EP_QKV) {
        f16 bv = (f16)val;
        if (n < 256) {                    // Q transposed: [t][h][d][s]
          int h = n >> 5, d = n & 31;
          qt[(((size_t)t * 8 + h) * 32 + d) * SD + m] = bv;
        } else if (n < 512) {             // K plain: [t][s][c]
          kb[(size_t)t * SD * CD + (size_t)m * CD + (n - 256)] = bv;
        } else {                          // V plain: [t][s][c]
          vb[(size_t)t * SD * CD + (size_t)m * CD + (n - 512)] = bv;
        }
      } else if constexpr (EP == EP_OUT) {
        int img = pidx[t];
        float g = ((const float*)(ws + OFF_GVAL))[t >> 1];
        float base = zsem[((size_t)img * CD + n) * SD + m];
        newx[(size_t)t * SD * CD + (size_t)m * CD + n] = base + g * val;
      } else if constexpr (EP == EP_MLP1) {
        float gl = 0.5f * val * (1.f + erff(val * 0.70710678118f));
        t1[(size_t)t * SD * 1024 + (size_t)m * 1024 + n] = (f16)gl;
      } else { // EP_MLP2: residual + transposed store into d_out
        int img = pidx[t];
        float nv = newx[(size_t)t * SD * CD + (size_t)m * CD + n] + val;
        dout[((size_t)img * CD + n) * SD + m] = nv;
      }
    }
  }
}

// ---------------- flash attention, transposed-score formulation -------------
// S^T = K_tile(16x32) @ Q^T(32x16): C-frag lane = query col, M = kv rows.
// Per-query softmax stats live in one lane (+xor-16 partner), and exp(S^T)
// packs directly into the A-fragment of the P @ V WMMA (identical mapping).
__global__ __launch_bounds__(256) void k_attn(const f16* __restrict__ qt,
                                              const f16* __restrict__ kb,
                                              const f16* __restrict__ vb,
                                              f16* __restrict__ ctx) {
  const int t = blockIdx.z, h = blockIdx.y;
  const int kvT = t ^ 1;                      // partner tensor supplies K/V
  const int w = threadIdx.x >> 5;
  const int lane = threadIdx.x & 31, half = lane >> 4, lm = lane & 15;
  const int q0 = blockIdx.x * 128 + w * 16;

  // Q^T B-frag, loop invariant: lane%16 = query column, d over elements
  Frag bq;
  const f16* qp = qt + ((size_t)t * 8 + h) * 32 * SD + q0 + lm;
#pragma unroll
  for (int s = 0; s < 16; ++s)
    bq.v[s] = qp[(size_t)(s < 8 ? 8 * half + s : 8 + 8 * half + s) * SD];

  v8f o0 = {}, o1 = {};
  float rm = -1e30f, rl = 0.f;                // stats for query q0+lm
  const f16* kbase = kb + (size_t)kvT * SD * CD + h * 32 + 8 * half;
  const f16* vbase = vb + (size_t)kvT * SD * CD + h * 32;
  const float scale = 0.17677669529663687f;   // 1/sqrt(32)

#pragma unroll 2
  for (int j0 = 0; j0 < SD; j0 += 32) {
    // K A-frags: lane row = kv j0+lm (and +16), contiguous d chunks
    Frag ka0, ka1;
    const f16* kp = kbase + (size_t)(j0 + lm) * CD;
    ka0.q[0] = *(const vf4*)kp;
    ka0.q[1] = *(const vf4*)(kp + 16);
    const f16* kp1 = kp + 16 * CD;
    ka1.q[0] = *(const vf4*)kp1;
    ka1.q[1] = *(const vf4*)(kp1 + 16);

    v8f s0 = {}, s1 = {};
    s0 = wmma16(ka0, bq, s0);  // kv rows j0 + i + 8*half
    s1 = wmma16(ka1, bq, s1);  // kv rows j0 + 16 + i + 8*half

    float x0[8], x1[8], ml = -1e30f;
#pragma unroll
    for (int i = 0; i < 8; ++i) {
      x0[i] = s0[i] * scale;
      x1[i] = s1[i] * scale;
      ml = fmaxf(ml, fmaxf(x0[i], x1[i]));
    }
    ml = fmaxf(ml, __shfl_xor(ml, 16, 32));   // other 16 kv rows of this tile
    float mn = fmaxf(rm, ml);
    float al = __expf(rm - mn);
    rm = mn;

    Frag ap;                                   // P packs straight into A-frag
    float ps = 0.f;
#pragma unroll
    for (int i = 0; i < 8; ++i) {
      float p0 = __expf(x0[i] - mn);
      float p1 = __expf(x1[i] - mn);
      ps += p0 + p1;
      ap.v[i]     = (f16)p0;                   // K-chunk [8*half, +8)
      ap.v[8 + i] = (f16)p1;                   // K-chunk [16+8*half, +8)
    }
    ps += __shfl_xor(ps, 16, 32);
    rl = al * rl + ps;

    // O rescale: alpha is per query; O rows are M = i + 8*half
#pragma unroll
    for (int i = 0; i < 8; ++i) {
      float alq = __shfl(al, i + 8 * half, 16);
      o0[i] *= alq;
      o1[i] *= alq;
    }

    // V B-frags: lane%16 = d column, kv rows over elements
    Frag vb0, vb1;
    const f16* vp = vbase + (size_t)j0 * CD + lm;
#pragma unroll
    for (int s = 0; s < 16; ++s) {
      int kr = s < 8 ? 8 * half + s : 8 + 8 * half + s;
      vb0.v[s] = vp[(size_t)kr * CD];
      vb1.v[s] = vp[(size_t)kr * CD + 16];
    }
    o0 = wmma16(ap, vb0, o0);
    o1 = wmma16(ap, vb1, o1);

    if (j0 + 32 < SD) {                        // latency hints for next tiles
      __builtin_prefetch(kp + 32 * CD, 0, 3);
      __builtin_prefetch(vp + 32 * CD, 0, 3);
    }
  }

  float inv = 1.f / rl;
#pragma unroll
  for (int i = 0; i < 8; ++i) {
    float invq = __shfl(inv, i + 8 * half, 16);
    int s = q0 + i + 8 * half;
    f16* cp = ctx + ((size_t)t * SD + s) * CD + h * 32;
    cp[lm]      = (f16)(o0[i] * invq);
    cp[16 + lm] = (f16)(o1[i] * invq);
  }
}

// ---------------- LN of token-major newx ----------------
__global__ __launch_bounds__(256) void k_ln2(const float* __restrict__ newx,
                                             const float* __restrict__ w,
                                             const float* __restrict__ b,
                                             f16* __restrict__ hln) {
  const int t = blockIdx.y;
  const int s = blockIdx.x * 8 + (threadIdx.x >> 5);
  const int lane = threadIdx.x & 31;
  const float* xp = newx + ((size_t)t * SD + s) * CD;
  float v[8]; float sum = 0.f, sq = 0.f;
#pragma unroll
  for (int k = 0; k < 8; ++k) {
    v[k] = xp[lane + 32 * k];
    sum += v[k]; sq += v[k] * v[k];
  }
  sum = wave_sum32(sum); sq = wave_sum32(sq);
  float mean = sum * (1.f / 256.f);
  float var  = sq * (1.f / 256.f) - mean * mean;
  float rstd = rsqrtf(var + 1e-5f);
  f16* op = hln + ((size_t)t * SD + s) * CD;
#pragma unroll
  for (int k = 0; k < 8; ++k) {
    int c = lane + 32 * k;
    op[c] = (f16)(((v[k] - mean) * rstd) * w[c] + b[c]);
  }
}

// ---------------- host launcher ----------------
extern "C" void kernel_launch(void* const* d_in, const int* in_sizes, int n_in,
                              void* d_out, int out_size, void* d_ws, size_t ws_size,
                              hipStream_t stream) {
  (void)in_sizes; (void)n_in; (void)out_size; (void)ws_size;
  const float* zsem = (const float*)d_in[0];
  const int*   pidx = (const int*)d_in[1];
  const float* lnw  = (const float*)d_in[2];
  const float* lnb  = (const float*)d_in[3];
  const float* inw  = (const float*)d_in[4];
  const float* inb  = (const float*)d_in[5];
  const float* outw = (const float*)d_in[6];
  const float* outb = (const float*)d_in[7];
  const float* res_scale  = (const float*)d_in[8];
  const float* gate_scale = (const float*)d_in[9];
  const float* mlnw = (const float*)d_in[10];
  const float* mlnb = (const float*)d_in[11];
  const float* w1   = (const float*)d_in[12];
  const float* b1   = (const float*)d_in[13];
  const float* w2   = (const float*)d_in[14];
  const float* b2   = (const float*)d_in[15];

  char* ws   = (char*)d_ws;
  float* out = (float*)d_out;
  f16* qln = (f16*)(ws + OFF_QLN);
  f16* qt  = (f16*)(ws + OFF_QT);
  f16* kb  = (f16*)(ws + OFF_KB);
  f16* vb  = (f16*)(ws + OFF_VB);
  f16* ctx = (f16*)(ws + OFF_CTX);
  float* newx = (float*)(ws + OFF_NEWX);
  f16* hln = (f16*)(ws + OFF_HLN);
  f16* t1  = (f16*)(ws + OFF_T1);

  k_prep<<<3072, 256, 0, stream>>>(inw, outw, w1, w2, ws);
  k_copy<<<4096, 256, 0, stream>>>(zsem, out);
  k_ln<<<dim3(32, 16), 256, 0, stream>>>(zsem, pidx, lnw, lnb, qln);
  k_mean<<<16, 256, 0, stream>>>(qln, (float*)(ws + OFF_MSUM));
  k_gate<<<8, 256, 0, stream>>>(ws, gate_scale, res_scale, out);
  k_gemm<EP_QKV, 1024, 768, 256><<<dim3(12, 8, 16), 256, 0, stream>>>(
      qln, (const f16*)(ws + OFF_WIN), inb, ws, nullptr, nullptr, nullptr);
  k_attn<<<dim3(8, 8, 16), 256, 0, stream>>>(qt, kb, vb, ctx);
  k_gemm<EP_OUT, 1024, 256, 256><<<dim3(4, 8, 16), 256, 0, stream>>>(
      ctx, (const f16*)(ws + OFF_WOUT), outb, ws, zsem, pidx, nullptr);
  k_ln2<<<dim3(128, 16), 256, 0, stream>>>(newx, mlnw, mlnb, hln);
  k_gemm<EP_MLP1, 1024, 1024, 256><<<dim3(16, 8, 16), 256, 0, stream>>>(
      hln, (const f16*)(ws + OFF_W1), b1, ws, nullptr, nullptr, nullptr);
  k_gemm<EP_MLP2, 1024, 256, 1024><<<dim3(4, 8, 16), 256, 0, stream>>>(
      t1, (const f16*)(ws + OFF_W2), b2, ws, nullptr, pidx, out);
}
